// Get_Correlation_28080496181852
// MI455X (gfx1250) — compile-verified, matrix operand order checked by reading, not run on hardware
//
#include <hip/hip_runtime.h>
#include <hip/hip_bf16.h>
#include <math.h>

// Problem constants
#define Bn   4
#define Cn   512
#define Tn   64
#define Hn   14
#define Wn   14
#define HWn  196
#define HWP  224   // 196 padded to 14*16; divisible by the K=32 WMMA step
#define RCn  32

typedef __bf16 bf16_t;
typedef __attribute__((ext_vector_type(16))) __bf16 v16bf;
typedef __attribute__((ext_vector_type(8)))  __bf16 v8bf;
typedef __attribute__((ext_vector_type(8)))  float  v8f;

// ---------------------------------------------------------------------------
// WMMA 16x16x32 bf16 fragment helpers (layouts per CDNA5 ISA 7.12.2, wave32)
//   A (16x32, MxK): lane l: M = l&15; hi = l>>4 selects K-halves.
//     element e<8  -> K = 8*hi + e ; element e>=8 -> K = 16 + 8*hi + (e-8)
//   B (32x16, KxN): mirror of A with N = l&15.
//   C/D (16x16 f32): vgpr v, lane l -> M = v + 8*(l>>4), N = l&15.
// All operands are per-lane CONTIGUOUS (two 16B chunks -> global_load_b128).
// Heavy kernels use 2x2 register blocking: 4 fragment loads feed 4 WMMAs.
// ---------------------------------------------------------------------------
__device__ __forceinline__ int frag_k(int e, int hi) {
  return (e < 8) ? (hi * 8 + e) : (16 + hi * 8 + (e - 8));
}

__device__ __forceinline__ v16bf ld_contig(const bf16_t* p, int hi) {
  v16bf v;
#pragma unroll
  for (int e = 0; e < 16; ++e) v[e] = p[frag_k(e, hi)];
  return v;
}

#define WMMA_BF16(ACC, A, B) \
  ACC = __builtin_amdgcn_wmma_f32_16x16x32_bf16(false, (A), false, (B), (short)0, (ACC), false, false)

__device__ __forceinline__ float sigm(float x) { return 1.0f / (1.0f + __expf(-x)); }

// ---------------------------------------------------------------------------
// 1) f32 -> bf16 conversion of x into TRANSPOSED padded layout [B,T,HWP,C]
//    (pad rows hw>=196 are zero so padded K/N regions contribute nothing)
// ---------------------------------------------------------------------------
__global__ void k_convert_x(const float* __restrict__ x, bf16_t* __restrict__ xbfT) {
  size_t i = (size_t)blockIdx.x * blockDim.x + threadIdx.x;
  const size_t total = (size_t)Bn * Tn * HWP * Cn;
  if (i >= total) return;
  int c  = (int)(i % Cn);  size_t r = i / Cn;
  int hw = (int)(r % HWP); r /= HWP;
  int t  = (int)(r % Tn);  int b = (int)(r / Tn);
  float v = 0.0f;
  if (hw < HWn) v = x[((size_t)(b * Cn + c) * Tn + t) * HWn + hw];
  xbfT[i] = (bf16_t)v;
}

__global__ void k_convert_w(const float* __restrict__ wdown2, const float* __restrict__ wdown,
                            const float* __restrict__ wback,
                            bf16_t* __restrict__ wd2, bf16_t* __restrict__ wd,
                            bf16_t* __restrict__ wb) {
  int i = blockIdx.x * blockDim.x + threadIdx.x;
  if (i < Cn * Cn)  wd2[i] = (bf16_t)wdown2[i];
  if (i < RCn * Cn) wd[i]  = (bf16_t)wdown[i];
  if (i < Cn * RCn) wb[i]  = (bf16_t)wback[i];
}

// ---------------------------------------------------------------------------
// 2) x2 = w_down2 @ x.  M=512, K=512, 2x2 blocked.
//    Writes x2bf [B,T,C,HWP] (features-A) and x2bfT [B,T,HWP,C] (aff-B).
// ---------------------------------------------------------------------------
__global__ void k_gemm_x2(const bf16_t* __restrict__ wd2, const bf16_t* __restrict__ xbfT,
                          bf16_t* __restrict__ x2bf, bf16_t* __restrict__ x2bfT) {
  int wid  = blockIdx.x * (blockDim.x >> 5) + (threadIdx.x >> 5);
  int lane = threadIdx.x & 31, hi = lane >> 4, ln = lane & 15;
  int np = wid % (HWP / 32); wid /= (HWP / 32);
  int t  = wid % Tn;         wid /= Tn;
  int mp = wid % (Cn / 32);  int b = wid / (Cn / 32);
  if (b >= Bn) return;  // wave-uniform

  const int m0 = mp * 32 + ln, n0 = np * 32 + ln, n1 = n0 + 16;
  const size_t tb = (size_t)(b * Tn + t) * HWP;
  const bf16_t* pa0 = wd2 + (size_t)m0 * Cn;
  const bf16_t* pa1 = pa0 + (size_t)16 * Cn;
  const bf16_t* pb0 = xbfT + (tb + n0) * Cn;
  const bf16_t* pb1 = xbfT + (tb + n1) * Cn;
  v8f a00 = {}, a01 = {}, a10 = {}, a11 = {};
#pragma unroll 4
  for (int kk = 0; kk < Cn / 32; ++kk) {
    int k0 = kk * 32;
    v16bf fa0 = ld_contig(pa0 + k0, hi);
    v16bf fa1 = ld_contig(pa1 + k0, hi);
    v16bf fb0 = ld_contig(pb0 + k0, hi);
    v16bf fb1 = ld_contig(pb1 + k0, hi);
    WMMA_BF16(a00, fa0, fb0); WMMA_BF16(a01, fa0, fb1);
    WMMA_BF16(a10, fa1, fb0); WMMA_BF16(a11, fa1, fb1);
  }
  const size_t ob = (size_t)(b * Tn + t) * Cn;
  v8bf p00, p01, p10, p11;
#pragma unroll
  for (int v = 0; v < 8; ++v) {
    int r0 = mp * 32 + v + 8 * hi, r1 = r0 + 16;
    x2bf[(ob + r0) * HWP + n0] = (bf16_t)a00[v];
    x2bf[(ob + r0) * HWP + n1] = (bf16_t)a01[v];
    x2bf[(ob + r1) * HWP + n0] = (bf16_t)a10[v];
    x2bf[(ob + r1) * HWP + n1] = (bf16_t)a11[v];
    p00[v] = (bf16_t)a00[v]; p01[v] = (bf16_t)a01[v];
    p10[v] = (bf16_t)a10[v]; p11[v] = (bf16_t)a11[v];
  }
  const int c0 = mp * 32 + 8 * hi, c1 = c0 + 16;   // 16B-aligned packed stores
  *(v8bf*)(x2bfT + (tb + n0) * Cn + c0) = p00;
  *(v8bf*)(x2bfT + (tb + n1) * Cn + c0) = p01;
  *(v8bf*)(x2bfT + (tb + n0) * Cn + c1) = p10;
  *(v8bf*)(x2bfT + (tb + n1) * Cn + c1) = p11;
}

// ---------------------------------------------------------------------------
// 3) xr = w_down @ x -> f32 [B,RC,T,HW].  M=32, K=512, 2x2 blocked over M.
// ---------------------------------------------------------------------------
__global__ void k_gemm_xr(const bf16_t* __restrict__ wd, const bf16_t* __restrict__ xbfT,
                          float* __restrict__ xr) {
  int wid  = blockIdx.x * (blockDim.x >> 5) + (threadIdx.x >> 5);
  int lane = threadIdx.x & 31, hi = lane >> 4, ln = lane & 15;
  int np = wid % (HWP / 32); wid /= (HWP / 32);
  int t  = wid % Tn;         int b = wid / Tn;   // M=32 = one m-pair
  if (b >= Bn) return;

  const int n0 = np * 32 + ln, n1 = n0 + 16;
  const size_t tb = (size_t)(b * Tn + t) * HWP;
  const bf16_t* pa0 = wd + (size_t)ln * Cn;
  const bf16_t* pa1 = pa0 + (size_t)16 * Cn;
  const bf16_t* pb0 = xbfT + (tb + n0) * Cn;
  const bf16_t* pb1 = xbfT + (tb + n1) * Cn;
  v8f a00 = {}, a01 = {}, a10 = {}, a11 = {};
#pragma unroll 4
  for (int kk = 0; kk < Cn / 32; ++kk) {
    int k0 = kk * 32;
    v16bf fa0 = ld_contig(pa0 + k0, hi);
    v16bf fa1 = ld_contig(pa1 + k0, hi);
    v16bf fb0 = ld_contig(pb0 + k0, hi);
    v16bf fb1 = ld_contig(pb1 + k0, hi);
    WMMA_BF16(a00, fa0, fb0); WMMA_BF16(a01, fa0, fb1);
    WMMA_BF16(a10, fa1, fb0); WMMA_BF16(a11, fa1, fb1);
  }
#pragma unroll
  for (int v = 0; v < 8; ++v) {
    int r0 = v + 8 * hi, r1 = r0 + 16;
    size_t rb0 = ((size_t)(b * RCn + r0) * Tn + t) * HWn;
    size_t rb1 = ((size_t)(b * RCn + r1) * Tn + t) * HWn;
    if (n0 < HWn) { xr[rb0 + n0] = a00[v]; xr[rb1 + n0] = a10[v]; }
    if (n1 < HWn) { xr[rb0 + n1] = a01[v]; xr[rb1 + n1] = a11[v]; }
  }
}

// ---------------------------------------------------------------------------
// 4) aff = x^T @ x2_shift per (b,t,dir), fused w2*(sigmoid-0.5), 2x2 blocked.
//    S stored in NATURAL [hw, sd] orientation -> features-B reads contiguous.
// ---------------------------------------------------------------------------
__global__ void k_aff(const bf16_t* __restrict__ xbfT, const bf16_t* __restrict__ x2bfT,
                      const float* __restrict__ weights2, bf16_t* __restrict__ S) {
  int wid  = blockIdx.x * (blockDim.x >> 5) + (threadIdx.x >> 5);
  int lane = threadIdx.x & 31, hi = lane >> 4, ln = lane & 15;
  int np = wid % (HWP / 32); wid /= (HWP / 32);
  int mp = wid % (HWP / 32); wid /= (HWP / 32);
  int t  = wid % Tn;         wid /= Tn;
  int b  = wid % Bn;         int dir = wid / Bn;
  if (dir >= 2) return;

  const int ts = (dir == 0) ? ((t + 1 < Tn) ? t + 1 : Tn - 1)
                            : ((t > 0) ? t - 1 : 0);
  const int m0 = mp * 32 + ln, n0 = np * 32 + ln, n1 = n0 + 16;
  const bf16_t* pa0 = xbfT  + ((size_t)(b * Tn + t)  * HWP + m0) * Cn;  // A[m=hw,k=c]
  const bf16_t* pa1 = pa0 + (size_t)16 * Cn;
  const bf16_t* pb0 = x2bfT + ((size_t)(b * Tn + ts) * HWP + n0) * Cn;  // B[k=c,n=sd]
  const bf16_t* pb1 = x2bfT + ((size_t)(b * Tn + ts) * HWP + n1) * Cn;
  v8f a00 = {}, a01 = {}, a10 = {}, a11 = {};
#pragma unroll 4
  for (int kk = 0; kk < Cn / 32; ++kk) {
    int k0 = kk * 32;
    __builtin_prefetch(pa0 + k0 + 128, 0, 0);   // global_prefetch_b8 on A stream
    v16bf fa0 = ld_contig(pa0 + k0, hi);
    v16bf fa1 = ld_contig(pa1 + k0, hi);
    v16bf fb0 = ld_contig(pb0 + k0, hi);
    v16bf fb1 = ld_contig(pb1 + k0, hi);
    WMMA_BF16(a00, fa0, fb0); WMMA_BF16(a01, fa0, fb1);
    WMMA_BF16(a10, fa1, fb0); WMMA_BF16(a11, fa1, fb1);
  }
  const float w2d = weights2[dir];
  bf16_t* sp = S + ((size_t)((dir * Bn + b) * Tn + t)) * HWP * HWP;
#pragma unroll
  for (int v = 0; v < 8; ++v) {
    int r0 = mp * 32 + v + 8 * hi, r1 = r0 + 16;   // hw index (row of S)
    sp[(size_t)r0 * HWP + n0] = (bf16_t)(w2d * (sigm(a00[v]) - 0.5f));
    sp[(size_t)r0 * HWP + n1] = (bf16_t)(w2d * (sigm(a01[v]) - 0.5f));
    sp[(size_t)r1 * HWP + n0] = (bf16_t)(w2d * (sigm(a10[v]) - 0.5f));
    sp[(size_t)r1 * HWP + n1] = (bf16_t)(w2d * (sigm(a11[v]) - 0.5f));
  }
}

// ---------------------------------------------------------------------------
// 5) features[c,hw] = sum_dir sum_sd x2_shift[c,sd] * S[hw,sd]; 2x2 blocked.
//    A from x2bf [.., c, sd] (contiguous in sd); B from S [hw, sd] (contig).
// ---------------------------------------------------------------------------
__global__ void k_features(const bf16_t* __restrict__ x2bf, const bf16_t* __restrict__ S,
                           float* __restrict__ features) {
  int wid  = blockIdx.x * (blockDim.x >> 5) + (threadIdx.x >> 5);
  int lane = threadIdx.x & 31, hi = lane >> 4, ln = lane & 15;
  int np = wid % (HWP / 32); wid /= (HWP / 32);
  int mp = wid % (Cn / 32);  wid /= (Cn / 32);
  int t  = wid % Tn;         int b = wid / Tn;
  if (b >= Bn) return;

  const int m0 = mp * 32 + ln, n0 = np * 32 + ln, n1 = n0 + 16;
  v8f a00 = {}, a01 = {}, a10 = {}, a11 = {};
#pragma unroll
  for (int dir = 0; dir < 2; ++dir) {
    const int ts = (dir == 0) ? ((t + 1 < Tn) ? t + 1 : Tn - 1)
                              : ((t > 0) ? t - 1 : 0);
    const bf16_t* pa0 = x2bf + ((size_t)(b * Tn + ts) * Cn + m0) * HWP;   // A[m=c,k=sd]
    const bf16_t* pa1 = pa0 + (size_t)16 * HWP;
    const bf16_t* sb = S + ((size_t)((dir * Bn + b) * Tn + t)) * HWP * HWP;
    const bf16_t* pb0 = sb + (size_t)n0 * HWP;                            // B[k=sd,n=hw]
    const bf16_t* pb1 = sb + (size_t)n1 * HWP;
#pragma unroll
    for (int kk = 0; kk < HWP / 32; ++kk) {
      int k0 = kk * 32;
      v16bf fa0 = ld_contig(pa0 + k0, hi);
      v16bf fa1 = ld_contig(pa1 + k0, hi);
      v16bf fb0 = ld_contig(pb0 + k0, hi);
      v16bf fb1 = ld_contig(pb1 + k0, hi);
      WMMA_BF16(a00, fa0, fb0); WMMA_BF16(a01, fa0, fb1);
      WMMA_BF16(a10, fa1, fb0); WMMA_BF16(a11, fa1, fb1);
    }
  }
#pragma unroll
  for (int v = 0; v < 8; ++v) {
    int r0 = mp * 32 + v + 8 * hi, r1 = r0 + 16;
    size_t f0 = ((size_t)(b * Cn + r0) * Tn + t) * HWn;
    size_t f1 = ((size_t)(b * Cn + r1) * Tn + t) * HWn;
    if (n0 < HWn) { features[f0 + n0] = a00[v]; features[f1 + n0] = a10[v]; }
    if (n1 < HWn) { features[f0 + n1] = a01[v]; features[f1 + n1] = a11[v]; }
  }
}

// ---------------------------------------------------------------------------
// 6) depthwise 9x3x3 convs, 3 dilations, weighted sum (tiny: 0.8 GFLOP).
//    Output transposed+bf16: aggrT [B,T,HWP,RC] so final-B is contiguous.
// ---------------------------------------------------------------------------
__global__ void k_dwconv(const float* __restrict__ xr,
                         const float* __restrict__ w1, const float* __restrict__ w2_,
                         const float* __restrict__ w3, const float* __restrict__ wts,
                         bf16_t* __restrict__ aggrT) {
  int i = blockIdx.x * blockDim.x + threadIdx.x;
  const int total = Bn * RCn * Tn * HWn;
  if (i >= total) return;
  int hw = i % HWn; int t = (i / HWn) % Tn;
  int rc = (i / (HWn * Tn)) % RCn; int b = i / (HWn * Tn * RCn);
  int h = hw / Wn, w = hw % Wn;
  const float* base = xr + (size_t)(b * RCn + rc) * Tn * HWn;
  float acc = 0.0f;
#pragma unroll
  for (int j = 0; j < 3; ++j) {
    const int d = j + 1;
    const float* wj = ((j == 0) ? w1 : (j == 1) ? w2_ : w3) + rc * 81;
    float a = 0.0f;
    for (int kt = 0; kt < 9; ++kt) {
      int tt = t + kt - 4;
      if ((unsigned)tt >= (unsigned)Tn) continue;
      const float* bt = base + (size_t)tt * HWn;
#pragma unroll
      for (int kh = 0; kh < 3; ++kh) {
        int hh = h + d * (kh - 1);
        if ((unsigned)hh >= (unsigned)Hn) continue;
#pragma unroll
        for (int kw = 0; kw < 3; ++kw) {
          int ww = w + d * (kw - 1);
          if ((unsigned)ww >= (unsigned)Wn) continue;
          a += wj[kt * 9 + kh * 3 + kw] * bt[hh * Wn + ww];
        }
      }
    }
    acc += wts[j] * a;
  }
  aggrT[((size_t)(b * Tn + t) * HWP + hw) * RCn + rc] = (bf16_t)acc;
}

// ---------------------------------------------------------------------------
// 7) agg = w_back @ aggr (K=32 -> single WMMA per tile), fused with
//    out = features * (sigmoid(agg) - 0.5)
// ---------------------------------------------------------------------------
__global__ void k_final(const bf16_t* __restrict__ wb, const bf16_t* __restrict__ aggrT,
                        const float* __restrict__ features, float* __restrict__ out) {
  int wid  = blockIdx.x * (blockDim.x >> 5) + (threadIdx.x >> 5);
  int lane = threadIdx.x & 31, hi = lane >> 4, ln = lane & 15;
  int np = wid % (HWP / 32); wid /= (HWP / 32);
  int mt = wid % (Cn / 16);  wid /= (Cn / 16);
  int t  = wid % Tn;         int b = wid / Tn;
  if (b >= Bn) return;

  const int m = mt * 16 + ln, n0 = np * 32 + ln, n1 = n0 + 16;
  const size_t tb = (size_t)(b * Tn + t) * HWP;
  v16bf a  = ld_contig(wb + (size_t)m * RCn, hi);        // A[m=c,k=rc], K=32 exactly
  v16bf b0 = ld_contig(aggrT + (tb + n0) * RCn, hi);     // B[k=rc,n=hw], contiguous
  v16bf b1 = ld_contig(aggrT + (tb + n1) * RCn, hi);
  v8f acc0 = {}, acc1 = {};
  WMMA_BF16(acc0, a, b0);
  WMMA_BF16(acc1, a, b1);
#pragma unroll
  for (int v = 0; v < 8; ++v) {
    int mo = mt * 16 + v + 8 * hi;
    size_t ob = ((size_t)(b * Cn + mo) * Tn + t) * HWn;
    if (n0 < HWn) out[ob + n0] = features[ob + n0] * (sigm(acc0[v]) - 0.5f);
    if (n1 < HWn) out[ob + n1] = features[ob + n1] * (sigm(acc1[v]) - 0.5f);
  }
}

// ---------------------------------------------------------------------------
// Launcher
// ---------------------------------------------------------------------------
extern "C" void kernel_launch(void* const* d_in, const int* in_sizes, int n_in,
                              void* d_out, int out_size, void* d_ws, size_t ws_size,
                              hipStream_t stream) {
  const float* x        = (const float*)d_in[0];
  const float* w_down   = (const float*)d_in[1];
  const float* w_down2  = (const float*)d_in[2];
  const float* w_sa1    = (const float*)d_in[3];
  const float* w_sa2    = (const float*)d_in[4];
  const float* w_sa3    = (const float*)d_in[5];
  const float* w_back   = (const float*)d_in[6];
  const float* weights  = (const float*)d_in[7];
  const float* weights2 = (const float*)d_in[8];
  float* out = (float*)d_out;

  char* ws = (char*)d_ws;
  const size_t n_xbf  = (size_t)Bn * Tn * HWP * Cn;            // bf16
  const size_t n_S    = (size_t)2 * Bn * Tn * HWP * HWP;       // bf16
  const size_t n_feat = (size_t)Bn * Cn * Tn * HWn;            // f32
  const size_t n_xr   = (size_t)Bn * RCn * Tn * HWn;           // f32
  const size_t n_aggT = (size_t)Bn * Tn * HWP * RCn;           // bf16

  bf16_t* xbfT  = (bf16_t*)ws;                 ws += n_xbf * 2;
  bf16_t* x2bf  = (bf16_t*)ws;                 ws += n_xbf * 2;
  bf16_t* x2bfT = (bf16_t*)ws;                 ws += n_xbf * 2;
  bf16_t* S     = (bf16_t*)ws;                 ws += n_S * 2;
  float*  feats = (float*)ws;                  ws += n_feat * 4;
  float*  xr    = (float*)ws;                  ws += n_xr * 4;
  bf16_t* aggrT = (bf16_t*)ws;                 ws += n_aggT * 2;
  bf16_t* wd2bf = (bf16_t*)ws;                 ws += (size_t)Cn * Cn * 2;
  bf16_t* wdbf  = (bf16_t*)ws;                 ws += (size_t)RCn * Cn * 2;
  bf16_t* wbbf  = (bf16_t*)ws;                 ws += (size_t)Cn * RCn * 2;

  const int blk = 256;                          // 8 wave32s per block
  // 1) conversions
  k_convert_w<<<(Cn * Cn + blk - 1) / blk, blk, 0, stream>>>(w_down2, w_down, w_back,
                                                             wd2bf, wdbf, wbbf);
  k_convert_x<<<(int)((n_xbf + blk - 1) / blk), blk, 0, stream>>>(x, xbfT);
  // 2) x2 GEMM: waves = B*(C/32)*T*(HWP/32) = 4*16*64*7 = 28672 -> 3584 blocks
  k_gemm_x2<<<(Bn * (Cn / 32) * Tn * (HWP / 32)) / 8, blk, 0, stream>>>(wd2bf, xbfT,
                                                                        x2bf, x2bfT);
  // 3) xr GEMM: waves = 4*64*7 = 1792 -> 224 blocks
  k_gemm_xr<<<(Bn * Tn * (HWP / 32)) / 8, blk, 0, stream>>>(wdbf, xbfT, xr);
  // 4) affinity + sigmoid: waves = 2*4*64*7*7 = 25088 -> 3136 blocks
  k_aff<<<(2 * Bn * Tn * (HWP / 32) * (HWP / 32)) / 8, blk, 0, stream>>>(xbfT, x2bfT,
                                                                         weights2, S);
  // 5) features: waves = 4*64*16*7 = 28672 -> 3584 blocks
  k_features<<<(Bn * Tn * (Cn / 32) * (HWP / 32)) / 8, blk, 0, stream>>>(x2bf, S, feats);
  // 6) depthwise convs
  k_dwconv<<<(Bn * RCn * Tn * HWn + blk - 1) / blk, blk, 0, stream>>>(xr, w_sa1, w_sa2,
                                                                      w_sa3, weights, aggrT);
  // 7) w_back GEMM + final elementwise: waves = 4*64*32*7 = 57344 -> 7168 blocks
  k_final<<<(Bn * Tn * (Cn / 16) * (HWP / 32)) / 8, blk, 0, stream>>>(wbbf, aggrT,
                                                                      feats, out);
}
